// CausalSelfAttention_38800734552778
// MI455X (gfx1250) — matrix operand order repaired and checked by reading, not compiled, not run on hardware
//
#include <hip/hip_runtime.h>
#include <stddef.h>

// ---------------------------------------------------------------------------
// Types for CDNA5 WMMA (wave32): V_WMMA_F32_16X16X32_BF16
// ---------------------------------------------------------------------------
typedef __attribute__((ext_vector_type(16))) __bf16       bf16x16;
typedef __attribute__((ext_vector_type(8)))  float        f32x8;
typedef __attribute__((ext_vector_type(4)))  unsigned int u32x4;

union Frag32B {           // 32 bytes = 16 bf16 = one A/B fragment per lane
    u32x4   q[2];
    bf16x16 v;
};

static __device__ __forceinline__ f32x8 zero8() {
    f32x8 z;
#pragma unroll
    for (int i = 0; i < 8; ++i) z[i] = 0.0f;
    return z;
}

static __device__ __forceinline__ f32x8 wmma_bf16(bf16x16 a, bf16x16 b, f32x8 c) {
    // (neg_a, A, neg_b, B, c_mod, C, reuse_a, reuse_b)
    return __builtin_amdgcn_wmma_f32_16x16x32_bf16(false, a, false, b, (short)0, c,
                                                   false, false);
}

static __device__ __forceinline__ unsigned short f2bf(float f) {
    union { float f; unsigned u; } v; v.f = f;
    unsigned u = v.u;
    unsigned rnd = 0x7FFFu + ((u >> 16) & 1u);   // round-to-nearest-even
    return (unsigned short)((u + rnd) >> 16);
}

static __device__ __forceinline__ u32x4 ld4(const unsigned short* p) {
    return *reinterpret_cast<const u32x4*>(p);   // 16B aligned by construction
}

// Two async global->LDS 16B copies (32B total). IOFFSET applies to BOTH the
// LDS and global addresses (ISA 08 §4.4), so one address pair serves both.
// Tracked with ASYNCcnt.
static __device__ __forceinline__ void async_ld32(unsigned ldsaddr,
                                                  unsigned long long gaddr) {
    asm volatile("global_load_async_to_lds_b128 %0, %1, off\n\t"
                 "global_load_async_to_lds_b128 %0, %1, off offset:16"
                 :
                 : "v"(ldsaddr), "v"(gaddr)
                 : "memory");
}
static __device__ __forceinline__ void wait_async0() {
    asm volatile("s_wait_asynccnt 0" ::: "memory");
}

// Problem constants
#define BATCH 2
#define TSEQ  2048
#define CDIM  1024
#define NHEAD 16
#define DHEAD 64
#define MROWS (BATCH * TSEQ)   // 4096
#define N3C   (3 * CDIM)       // 3072

// ---------------------------------------------------------------------------
// fp32 -> bf16 elementwise convert
// ---------------------------------------------------------------------------
__global__ void k_cvt(const float* __restrict__ s, unsigned short* __restrict__ d, int n) {
    int i = blockIdx.x * blockDim.x + threadIdx.x;
    if (i < n) d[i] = f2bf(s[i]);
}

// fp32 [K][N] -> bf16 [N][K] tiled transpose-convert (K, N multiples of 32)
__global__ void k_transpose_cvt(const float* __restrict__ src,
                                unsigned short* __restrict__ dst, int K, int N) {
    __shared__ unsigned short tile[32][33];
    const int kb = blockIdx.y * 32, nb = blockIdx.x * 32;
#pragma unroll
    for (int i = threadIdx.y; i < 32; i += 8)
        tile[i][threadIdx.x] = f2bf(src[(size_t)(kb + i) * N + nb + threadIdx.x]);
    __syncthreads();
#pragma unroll
    for (int i = threadIdx.y; i < 32; i += 8)
        dst[(size_t)(nb + i) * K + kb + threadIdx.x] = tile[threadIdx.x][i];
}

// ---------------------------------------------------------------------------
// GEMM core: block tile 256(M) x 64(N), K = 1024, A row-major [M][K] bf16,
// B = W^T [N][K] bf16. 8 waves; wave w computes rows m0+32w..+31 (two 16-row
// A fragments, so each LDS B fragment feeds 2 WMMAs). B tile staged into
// double-buffered LDS by async global->LDS loads; one barrier per k-step.
// A-frag per-lane layout used here (same order as B): lanes 0-15 hold K 0..15,
// lanes 16-31 hold K 16..31. B per-lane: lanes 0-15 K 0..15, lanes
// 16-31 K 16..31, N = lane%16.
// ---------------------------------------------------------------------------
static __device__ __forceinline__ void gemm_core(
    const unsigned short* __restrict__ Amat,   // [M][1024]
    const unsigned short* __restrict__ wt,     // [N][1024]
    unsigned short* Bl,                        // shared: 2 buffers x 2048 ushort
    int m0, int n0, f32x8 acc[2][4]) {
    const int tid = threadIdx.x, wave = tid >> 5, lane = tid & 31;
    const int nl = lane & 15, hi = lane >> 4;
    const int abase = hi ? 16 : 0;
    const unsigned short* ar0 = Amat + (size_t)(m0 + wave * 32 + nl) * CDIM;
    const unsigned short* ar1 = ar0 + (size_t)16 * CDIM;

#pragma unroll
    for (int mt = 0; mt < 2; ++mt)
#pragma unroll
        for (int g = 0; g < 4; ++g) acc[mt][g] = zero8();

    // producer setup: threads 0..127 stage the 32x64 B tile (fragment layout)
    const bool prod = tid < 128;
    unsigned ldsA = 0;
    unsigned long long gsrc = 0;
    if (prod) {
        const int ng = tid >> 5, L = tid & 31, nls = L & 15;
        const int kb16 = (L >> 4) ? 16 : 0;
        gsrc = (unsigned long long)(wt + (size_t)(n0 + ng * 16 + nls) * CDIM + kb16);
        ldsA = (unsigned)(unsigned long long)(Bl + (ng * 32 + L) * 16);
    }

    // prologue: stage k0=0 into buffer 0
    if (prod) async_ld32(ldsA, gsrc);
    wait_async0();
    __syncthreads();

    int buf = 0;
    for (int k0 = 0; k0 < CDIM; k0 += 32) {
        if (prod && (k0 + 32 < CDIM))
            async_ld32(ldsA + (unsigned)((buf ^ 1) * 4096),
                       gsrc + (unsigned long long)((k0 + 32) * 2));

        Frag32B af0, af1;
        af0.q[0] = ld4(ar0 + k0 + abase);
        af0.q[1] = ld4(ar0 + k0 + 8 + abase);
        af1.q[0] = ld4(ar1 + k0 + abase);
        af1.q[1] = ld4(ar1 + k0 + 8 + abase);

        const unsigned short* bbuf = Bl + buf * 2048;
#pragma unroll
        for (int g = 0; g < 4; ++g) {
            Frag32B bf;
            const unsigned short* p = bbuf + (g * 32 + lane) * 16;
            bf.q[0] = *reinterpret_cast<const u32x4*>(p);
            bf.q[1] = *reinterpret_cast<const u32x4*>(p + 8);
            acc[0][g] = wmma_bf16(af0.v, bf.v, acc[0][g]);
            acc[1][g] = wmma_bf16(af1.v, bf.v, acc[1][g]);
        }
        wait_async0();
        __syncthreads();
        buf ^= 1;
    }
}

// QKV projection: epilogue adds bias, pre-scales Q by 1/sqrt(D), splits into
// Q[bh][t][d], K[bh][t][d], V^T[bh][d][t]
__global__ __launch_bounds__(256) void k_gemm_qkv(
    const unsigned short* __restrict__ xb,   // [4096][1024] bf16
    const unsigned short* __restrict__ wt,   // [3072][1024] bf16 (Wqkv^T)
    const float* __restrict__ bias,          // [3072] f32
    unsigned short* __restrict__ qg,
    unsigned short* __restrict__ kg,
    unsigned short* __restrict__ vtg) {
    __shared__ __align__(16) unsigned short Bl[2 * 2048];   // 8 KB
    const int m0 = blockIdx.x * 256, n0 = blockIdx.y * 64;
    const int tid = threadIdx.x, wave = tid >> 5, lane = tid & 31;
    const int nl = lane & 15, hi = lane >> 4;

    f32x8 acc[2][4];
    gemm_core(xb, wt, Bl, m0, n0, acc);

    // C/D layout: lane L: N = L%16; VGPR r holds M = r + 8*(L>=16).
#pragma unroll
    for (int mt = 0; mt < 2; ++mt) {
#pragma unroll
        for (int g = 0; g < 4; ++g) {
            const int nc = n0 + g * 16 + nl;
            const float bv = bias[nc];
            const int sect = nc >> 10;     // uniform per block (64 | 1024)
            const int cc = nc & 1023;
            const int hh = cc >> 6, dd = cc & 63;
#pragma unroll
            for (int r = 0; r < 8; ++r) {
                const int row = m0 + wave * 32 + mt * 16 + r + hi * 8;
                const int bb = row >> 11, tt = row & 2047;
                const int bh = bb * NHEAD + hh;
                float val = acc[mt][g][r] + bv;
                if (sect == 0) {
                    val *= 0.125f;   // fold 1/sqrt(64) into Q
                    qg[((size_t)bh * TSEQ + tt) * DHEAD + dd] = f2bf(val);
                } else if (sect == 1) {
                    kg[((size_t)bh * TSEQ + tt) * DHEAD + dd] = f2bf(val);
                } else {
                    vtg[((size_t)bh * DHEAD + dd) * TSEQ + tt] = f2bf(val);
                }
            }
        }
    }
}

// Output projection: fp32 output + bias
__global__ __launch_bounds__(256) void k_gemm_proj(
    const unsigned short* __restrict__ yb,   // [4096][1024] bf16
    const unsigned short* __restrict__ wt,   // [1024][1024] bf16 (Wproj^T)
    const float* __restrict__ bias,          // [1024]
    float* __restrict__ out) {               // [4096][1024] f32
    __shared__ __align__(16) unsigned short Bl[2 * 2048];
    const int m0 = blockIdx.x * 256, n0 = blockIdx.y * 64;
    const int tid = threadIdx.x, wave = tid >> 5, lane = tid & 31;
    const int nl = lane & 15, hi = lane >> 4;

    f32x8 acc[2][4];
    gemm_core(yb, wt, Bl, m0, n0, acc);

#pragma unroll
    for (int mt = 0; mt < 2; ++mt) {
#pragma unroll
        for (int g = 0; g < 4; ++g) {
            const int nc = n0 + g * 16 + nl;
            const float bv = bias[nc];
#pragma unroll
            for (int r = 0; r < 8; ++r) {
                const int row = m0 + wave * 32 + mt * 16 + r + hi * 8;
                out[(size_t)row * CDIM + nc] = acc[mt][g][r] + bv;
            }
        }
    }
}

// ---------------------------------------------------------------------------
// Flash attention. grid = (B*H, T/256). Each of the 8 waves owns TWO 16-row
// query tiles (32 q rows) and streams 32-key chunks with online softmax:
// every K/V fragment load now feeds 2 WMMAs. No block barriers (waves have
// independent causal ranges). Q arrives pre-scaled by 1/sqrt(D).
// ---------------------------------------------------------------------------
__global__ __launch_bounds__(256) void k_attn(
    const unsigned short* __restrict__ qg,   // [BH][T][D] bf16 (pre-scaled)
    const unsigned short* __restrict__ kg,   // [BH][T][D] bf16
    const unsigned short* __restrict__ vtg,  // [BH][D][T] bf16
    unsigned short* __restrict__ yg) {       // [B*T][C] bf16
    __shared__ __align__(16) unsigned short plds[8 * 2 * 512];   // 16 KB

    const int bh = blockIdx.x;        // 0..31
    const int qt = blockIdx.y;        // 0..7
    const int wave = threadIdx.x >> 5;
    const int lane = threadIdx.x & 31;
    const int nl = lane & 15, hi = lane >> 4;
    const int base = qt * 256 + wave * 32;
    const int b = bh >> 4, h = bh & 15;
    const int abase = hi ? 8 : 0;
    const int kb16 = hi ? 16 : 0;
    const float NEG = -3.0e38f;

    // Q fragments (A layout) for both 16-row tiles, d-chunks [0,32) / [32,64)
    Frag32B qf[2][2];
#pragma unroll
    for (int mt = 0; mt < 2; ++mt) {
        const unsigned short* qrow =
            qg + ((size_t)bh * TSEQ + (base + mt * 16 + nl)) * DHEAD;
#pragma unroll
        for (int dg = 0; dg < 2; ++dg) {
            qf[mt][dg].q[0] = ld4(qrow + dg * 32 + kb16);
            qf[mt][dg].q[1] = ld4(qrow + dg * 32 + kb16 + 8);
        }
    }

    float m_run[2][8], l_run[2][8];
    f32x8 o[2][4];
#pragma unroll
    for (int mt = 0; mt < 2; ++mt) {
#pragma unroll
        for (int r = 0; r < 8; ++r) { m_run[mt][r] = NEG; l_run[mt][r] = 0.0f; }
#pragma unroll
        for (int g = 0; g < 4; ++g) o[mt][g] = zero8();
    }

    const int qmax = base + 31;
    const int nchunk = (qmax >> 5) + 1;

    for (int c = 0; c < nchunk; ++c) {
        const int key0 = c << 5;

        // K fragments for this chunk (shared by both q tiles)
        Frag32B kf[2][2];
#pragma unroll
        for (int st = 0; st < 2; ++st) {
            const unsigned short* krow =
                kg + ((size_t)bh * TSEQ + (key0 + st * 16 + nl)) * DHEAD;
            kf[st][0].q[0] = ld4(krow + kb16);
            kf[st][0].q[1] = ld4(krow + kb16 + 8);
            kf[st][1].q[0] = ld4(krow + 32 + kb16);
            kf[st][1].q[1] = ld4(krow + 32 + kb16 + 8);
        }

#pragma unroll
        for (int mt = 0; mt < 2; ++mt) {
            // S = Q K^T, two 16-key subtiles
            f32x8 s[2];
#pragma unroll
            for (int st = 0; st < 2; ++st) {
                f32x8 a = wmma_bf16(qf[mt][0].v, kf[st][0].v, zero8());
                a = wmma_bf16(qf[mt][1].v, kf[st][1].v, a);
                s[st] = a;
            }
            // causal mask (scale pre-folded into Q)
#pragma unroll
            for (int st = 0; st < 2; ++st) {
                const int keyn = key0 + st * 16 + nl;
#pragma unroll
                for (int r = 0; r < 8; ++r) {
                    const int qm = base + mt * 16 + r + hi * 8;
                    s[st][r] = (keyn <= qm) ? s[st][r] : NEG;
                }
            }
            // online softmax (reduce across the 16-lane N group)
            float mnew[8], corr[8];
#pragma unroll
            for (int r = 0; r < 8; ++r) {
                float rm = fmaxf(s[0][r], s[1][r]);
                for (int off = 1; off < 16; off <<= 1)
                    rm = fmaxf(rm, __shfl_xor(rm, off, 32));
                const float mn = fmaxf(m_run[mt][r], rm);
                mnew[r] = mn;
                corr[r] = __expf(m_run[mt][r] - mn);
            }
#pragma unroll
            for (int st = 0; st < 2; ++st)
#pragma unroll
                for (int r = 0; r < 8; ++r)
                    s[st][r] = __expf(s[st][r] - mnew[r]);
#pragma unroll
            for (int r = 0; r < 8; ++r) {
                float rs = s[0][r] + s[1][r];
                for (int off = 1; off < 16; off <<= 1)
                    rs += __shfl_xor(rs, off, 32);
                l_run[mt][r] = l_run[mt][r] * corr[r] + rs;
                m_run[mt][r] = mnew[r];
            }
#pragma unroll
            for (int g = 0; g < 4; ++g)
#pragma unroll
                for (int r = 0; r < 8; ++r) o[mt][g][r] *= corr[r];

            // P: C-layout -> A-fragment layout via wave-private LDS slab.
            // Element (m,kk): lane = m + 16*((kk>>3)&1), j = (kk&7) + 8*(kk>>4).
            unsigned short* myp = plds + (wave * 2 + mt) * 512;
            const int half = (nl >> 3) & 1;
#pragma unroll
            for (int st = 0; st < 2; ++st) {
                const int j = (nl & 7) + st * 8;
#pragma unroll
                for (int r = 0; r < 8; ++r) {
                    const int ml = r + hi * 8;
                    myp[(ml + 16 * half) * 16 + j] = f2bf(s[st][r]);
                }
            }
        }

        asm volatile("s_wait_dscnt 0" ::: "memory");   // wave-internal RAW on LDS
        Frag32B pf[2];
#pragma unroll
        for (int mt = 0; mt < 2; ++mt) {
            const unsigned short* myp = plds + (wave * 2 + mt) * 512;
            pf[mt].q[0] = *reinterpret_cast<const u32x4*>(myp + lane * 16);
            pf[mt].q[1] = *reinterpret_cast<const u32x4*>(myp + lane * 16 + 8);
        }

        // O += P @ V : each V fragment (N=d, K=key, key-contiguous via V^T)
        // feeds both q tiles.
#pragma unroll
        for (int g = 0; g < 4; ++g) {
            const unsigned short* vrow =
                vtg + ((size_t)bh * DHEAD + (g * 16 + nl)) * TSEQ + key0 + abase;
            Frag32B vf;
            vf.q[0] = ld4(vrow);
            vf.q[1] = ld4(vrow + 16);
            o[0][g] = wmma_bf16(pf[0].v, vf.v, o[0][g]);
            o[1][g] = wmma_bf16(pf[1].v, vf.v, o[1][g]);
        }
    }

    // normalize and store y[b*T + t][h*64 + d] as bf16
#pragma unroll
    for (int mt = 0; mt < 2; ++mt) {
        float inv[8];
#pragma unroll
        for (int r = 0; r < 8; ++r) inv[r] = 1.0f / l_run[mt][r];
#pragma unroll
        for (int g = 0; g < 4; ++g) {
#pragma unroll
            for (int r = 0; r < 8; ++r) {
                const int row = b * TSEQ + base + mt * 16 + r + hi * 8;
                const int col = h * DHEAD + g * 16 + nl;
                yg[(size_t)row * CDIM + col] = f2bf(o[mt][g][r] * inv[r]);
            }
        }
    }
}

// ---------------------------------------------------------------------------
// Launch
// ---------------------------------------------------------------------------
extern "C" void kernel_launch(void* const* d_in, const int* in_sizes, int n_in,
                              void* d_out, int out_size, void* d_ws, size_t ws_size,
                              hipStream_t stream) {
    const float* x     = (const float*)d_in[0];   // [2][2048][1024]
    const float* Wqkv  = (const float*)d_in[1];   // [1024][3072]
    const float* bqkv  = (const float*)d_in[2];   // [3072]
    const float* Wproj = (const float*)d_in[3];   // [1024][1024]
    const float* bproj = (const float*)d_in[4];   // [1024]
    float* out = (float*)d_out;                   // [4096][1024]

    // Workspace carve (~48 MB)
    size_t off = 0;
    auto carve = [&](size_t elems) -> unsigned short* {
        unsigned short* p = (unsigned short*)((char*)d_ws + off);
        off += elems * sizeof(unsigned short);
        off = (off + 255) & ~(size_t)255;
        return p;
    };
    unsigned short* xb  = carve((size_t)MROWS * CDIM);      // x bf16
    unsigned short* w1t = carve((size_t)N3C * CDIM);        // Wqkv^T bf16
    unsigned short* w2t = carve((size_t)CDIM * CDIM);       // Wproj^T bf16
    unsigned short* qg  = carve((size_t)32 * TSEQ * DHEAD); // Q  [bh][t][d]
    unsigned short* kg  = carve((size_t)32 * TSEQ * DHEAD); // K  [bh][t][d]
    unsigned short* vtg = carve((size_t)32 * DHEAD * TSEQ); // V^T[bh][d][t]
    unsigned short* yg  = carve((size_t)MROWS * CDIM);      // attn out bf16
    if (off > ws_size) return;  // workspace too small: cannot run safely

    // 1) precision conversion (+ weight transposes for K-contiguous B frags)
    {
        const int n = MROWS * CDIM;
        k_cvt<<<dim3((n + 255) / 256), dim3(256), 0, stream>>>(x, xb, n);
        k_transpose_cvt<<<dim3(N3C / 32, CDIM / 32), dim3(32, 8), 0, stream>>>(
            Wqkv, w1t, CDIM, N3C);
        k_transpose_cvt<<<dim3(CDIM / 32, CDIM / 32), dim3(32, 8), 0, stream>>>(
            Wproj, w2t, CDIM, CDIM);
    }
    // 2) QKV projection (bf16 WMMA, async-LDS double-buffered B staging)
    k_gemm_qkv<<<dim3(MROWS / 256, N3C / 64), dim3(256), 0, stream>>>(
        xb, w1t, bqkv, qg, kg, vtg);
    // 3) flash attention (bf16 WMMA, online softmax, 32 q-rows per wave)
    k_attn<<<dim3(BATCH * NHEAD, TSEQ / 256), dim3(256), 0, stream>>>(qg, kg, vtg, yg);
    // 4) output projection, fp32 result + bias
    k_gemm_proj<<<dim3(MROWS / 256, CDIM / 64), dim3(256), 0, stream>>>(
        yg, w2t, bproj, out);
}
